// MultiHeadBatchedMixers_75007308857793
// MI455X (gfx1250) — matrix operand order repaired and checked by reading, not compiled, not run on hardware
//
#include <hip/hip_runtime.h>
#include <hip/hip_bf16.h>

// ---- problem constants ----
constexpr int E_  = 8;
constexpr int H_  = 12;
constexpr int N_  = 256;   // tokens
constexpr int HD_ = 64;    // head dim
constexpr int HID_= 512;   // hidden
constexpr int B_  = 32;
constexpr int K_  = 2;
constexpr int HT  = 64;            // hid chunk size
constexpr int NCH = HID_ / HT;     // 8 chunks

typedef __attribute__((ext_vector_type(16))) __bf16 v16bf;
typedef __attribute__((ext_vector_type(4)))  __bf16 v4bf;
typedef __attribute__((ext_vector_type(8)))  float  v8f;

__device__ __forceinline__ v8f v8zero() {
  v8f z;
#pragma unroll
  for (int i = 0; i < 8; ++i) z[i] = 0.0f;
  return z;
}

__device__ __forceinline__ v8f wmma_bf16(v16bf a, v16bf b, v8f c) {
  return __builtin_amdgcn_wmma_f32_16x16x32_bf16(
      false, a, false, b, (short)0, c, false, false);
}

// A fragment, row-major [row][k]: lane-half pattern is two contiguous 8-elem runs
__device__ __forceinline__ v16bf load_a(const __bf16* src, int ld, int row,
                                        int kbase, int half) {
  const __bf16* p = src + row * ld + kbase + half * 8;
  v16bf a;
#pragma unroll
  for (int j = 0; j < 8; ++j) a[j] = p[j];
#pragma unroll
  for (int j = 0; j < 8; ++j) a[8 + j] = p[16 + j];
  return a;
}

// B fragment with B[k][col] stored transposed: src[col*ld + k] (two contiguous 8-elem runs)
__device__ __forceinline__ v16bf load_b_t(const __bf16* src, int ld, int kbase,
                                          int half, int col) {
  const __bf16* p = src + col * ld + kbase + half * 8;
  v16bf b;
#pragma unroll
  for (int j = 0; j < 8; ++j) b[j] = p[j];
#pragma unroll
  for (int j = 0; j < 8; ++j) b[8 + j] = p[16 + j];
  return b;
}

__device__ __forceinline__ float gelu_tanh(float v) {
  const float c0 = 0.7978845608028654f;  // sqrt(2/pi)
  const float c1 = 0.044715f;
  float u = c0 * (v + c1 * v * v * v);
  return 0.5f * v * (1.0f + tanhf(u));
}

// contiguous f32 -> bf16 copy: 64 elements per thread, float4 / v4bf vectorized
__device__ __forceinline__ void cvt_copy64(const float* __restrict__ src,
                                           __bf16* __restrict__ dst) {
#pragma unroll
  for (int j = 0; j < 16; ++j) {
    float4 v = ((const float4*)src)[j];
    v4bf o;
    o[0] = (__bf16)v.x; o[1] = (__bf16)v.y;
    o[2] = (__bf16)v.z; o[3] = (__bf16)v.w;
    ((v4bf*)dst)[j] = o;
  }
}

// ---- CDNA5 async global->LDS (GVS mode: saddr base + 32-bit vgpr offset) ----
__device__ __forceinline__ uint32_t lds_addr_of(const void* p) {
  return (uint32_t)(size_t)(__attribute__((address_space(3))) const void*)p;
}
__device__ __forceinline__ void async_b128(uint32_t lds_off, uint64_t gbase,
                                           uint32_t goff) {
  asm volatile("global_load_async_to_lds_b128 %0, %1, %2"
               :: "v"(lds_off), "v"(goff), "s"(gbase) : "memory");
}
__device__ __forceinline__ void wait_asynccnt0() {
  asm volatile("s_wait_asynccnt 0" ::: "memory");
}

// pre-pass: f32 -> bf16 (vectorized), n multiple of 4
__global__ __launch_bounds__(256)
void cvt_bf16_kernel(const float* __restrict__ src, __bf16* __restrict__ dst,
                     int n4) {
  int i = blockIdx.x * blockDim.x + threadIdx.x;
  int stride = gridDim.x * blockDim.x;
  for (; i < n4; i += stride) {
    float4 v = ((const float4*)src)[i];
    v4bf o;
    o[0] = (__bf16)v.x; o[1] = (__bf16)v.y;
    o[2] = (__bf16)v.z; o[3] = (__bf16)v.w;
    ((v4bf*)dst)[i] = o;
  }
}

template <bool WS>
__global__ __launch_bounds__(256, 1)
void moe_mixer_kernel(const void* __restrict__ xg_v,
                      const int*  __restrict__ eidx,
                      const float* __restrict__ ewt,
                      const void* __restrict__ w1t_v,
                      const float* __restrict__ w1c_all,
                      const float* __restrict__ b1_all,
                      const void* __restrict__ w2t_v,
                      const float* __restrict__ w2c_all,
                      const float* __restrict__ b2_all,
                      float* __restrict__ out) {
  // ---- LDS: 96 KB (CDNA5 WGP: 320 KB) ----
  __shared__ __bf16 xs[HD_ * N_];     // 32 KB  x bf16, [d][n]
  __shared__ __bf16 stage[HD_ * N_];  // 32 KB  w1t chunk [hid_l][n] / w2t chunk [n][hid_l] / z^T [n][d]
  __shared__ __bf16 ys[HT * HD_];     //  8 KB  y^T chunk [hid_l][d]
  __shared__ __bf16 hs[HD_ * HT];     //  8 KB  gelu chunk [d][hid_l]
  __shared__ __bf16 w1cs[HD_ * HD_];  //  8 KB
  __shared__ __bf16 w2cs[HD_ * HD_];  //  8 KB

  const int bh   = blockIdx.x;
  const int h    = bh % H_;
  const int tid  = threadIdx.x;
  const int wave = tid >> 5;
  const int lane = tid & 31;
  const int half = lane >> 4;
  const int l16  = lane & 15;

  // ---- load x once (64x256) ----
  if constexpr (WS) {
    const __bf16* xgb = (const __bf16*)xg_v + (size_t)bh * HD_ * N_;
    uint64_t gb = (uint64_t)xgb;
    uint32_t ls = lds_addr_of(xs);
#pragma unroll
    for (int j = 0; j < 8; ++j) {
      uint32_t off = (uint32_t)(j * 256 + tid) * 16u;
      async_b128(ls + off, gb, off);
    }
    wait_asynccnt0();
  } else {
    const float* xg = (const float*)xg_v + (size_t)bh * HD_ * N_;
    cvt_copy64(xg + tid * 64, xs + tid * 64);
  }

  v8f outacc[4][2];
#pragma unroll
  for (int dt = 0; dt < 4; ++dt)
#pragma unroll
    for (int t = 0; t < 2; ++t) outacc[dt][t] = v8zero();

  for (int k = 0; k < K_; ++k) {
    const int   e  = eidx[bh * K_ + k];
    const float wk = ewt[bh * K_ + k];
    const size_t eh = (size_t)e * H_ + h;
    const float* w1cg = w1c_all + eh * (size_t)(HD_ * HD_);
    const float* b1g  = b1_all  + eh * (size_t)(HD_ * HID_);
    const float* w2cg = w2c_all + eh * (size_t)(HD_ * HD_);
    const float* b2g  = b2_all  + eh * (size_t)(HD_ * N_);

    // channel-mix weights -> LDS bf16 (64 contiguous elements / thread)
    cvt_copy64(w1cg + (tid & 63) * 64, w1cs + (tid & 63) * 64);  // covered 4x; cheap & simple
    cvt_copy64(w2cg + (tid & 63) * 64, w2cs + (tid & 63) * 64);

    v8f zacc[4][2];
#pragma unroll
    for (int dt = 0; dt < 4; ++dt)
#pragma unroll
      for (int t = 0; t < 2; ++t) zacc[dt][t] = v8zero();

    for (int c = 0; c < NCH; ++c) {
      const int hid0 = c * HT;
      __syncthreads();  // previous users of stage/ys/hs done

      // ---- stage = W1t chunk [hid_l][n] (contiguous 32 KB) ----
      if constexpr (WS) {
        const __bf16* w1tgb = (const __bf16*)w1t_v + eh * (size_t)(HID_ * N_)
                              + (size_t)hid0 * N_;
        uint64_t gb = (uint64_t)w1tgb;
        uint32_t ls = lds_addr_of(stage);
#pragma unroll
        for (int j = 0; j < 8; ++j) {
          uint32_t off = (uint32_t)(j * 256 + tid) * 16u;
          async_b128(ls + off, gb, off);
        }
        wait_asynccnt0();
      } else {
        const float* w1tg = (const float*)w1t_v + eh * (size_t)(HID_ * N_);
        cvt_copy64(w1tg + (size_t)hid0 * N_ + tid * 64, stage + tid * 64);
        if (c + 1 < NCH)
          __builtin_prefetch(w1tg + (size_t)(hid0 + HT) * N_, 0, 1);
      }
      __syncthreads();

      // ---- GEMM1: y_c(64x64) = x(64x256) @ W1t_chunk^T; K=256; y stored transposed ----
      for (int t = 0; t < 2; ++t) {
        int id = wave * 2 + t, dt = id >> 2, ht = id & 3;
        v8f acc = v8zero();
#pragma unroll
        for (int kc = 0; kc < N_ / 32; ++kc) {
          v16bf a = load_a(xs, N_, dt * 16 + l16, kc * 32, half);
          v16bf b = load_b_t(stage, N_, kc * 32, half, ht * 16 + l16);
          acc = wmma_bf16(a, b, acc);
        }
#pragma unroll
        for (int i = 0; i < 8; ++i)  // y^T[hid_l][d]
          ys[(ht * 16 + l16) * HD_ + dt * 16 + half * 8 + i] = (__bf16)acc[i];
      }
      __syncthreads();

      // ---- stage = W2t chunk [n][hid_l]; async-overlapped with GEMM2 ----
      if constexpr (WS) {
        const __bf16* w2tgb = (const __bf16*)w2t_v + eh * (size_t)(N_ * HID_) + hid0;
        uint64_t gb = (uint64_t)w2tgb;
        uint32_t ls = lds_addr_of(stage);
#pragma unroll
        for (int j = 0; j < 8; ++j) {
          int cid = j * 256 + tid;
          int n = cid >> 3, sub = cid & 7;
          async_b128(ls + (uint32_t)cid * 16u, gb,
                     (uint32_t)(n * HID_ * 2 + sub * 16));
        }
      } else {
        const float* w2tg = (const float*)w2t_v + eh * (size_t)(N_ * HID_);
        cvt_copy64(w2tg + (size_t)tid * HID_ + hid0, stage + tid * 64);
      }

      // ---- GEMM2: h_c = GELU(W1c @ y_c + b1); K=64 ----
      for (int t = 0; t < 2; ++t) {
        int id = wave * 2 + t, dt = id >> 2, ht = id & 3;
        v8f acc = v8zero();
#pragma unroll
        for (int kc = 0; kc < 2; ++kc) {
          v16bf a = load_a(w1cs, HD_, dt * 16 + l16, kc * 32, half);
          v16bf b = load_b_t(ys, HD_, kc * 32, half, ht * 16 + l16);
          acc = wmma_bf16(a, b, acc);
        }
#pragma unroll
        for (int i = 0; i < 8; ++i) {
          int m = dt * 16 + half * 8 + i, nn = ht * 16 + l16;
          float v = acc[i] + b1g[(size_t)m * HID_ + hid0 + nn];
          hs[m * HT + nn] = (__bf16)gelu_tanh(v);
        }
      }
      if constexpr (WS) wait_asynccnt0();
      __syncthreads();

      // ---- GEMM3: z(64x256) += h_c(64x64) @ W2t_chunk^T; K=64 ----
      {
        v16bf b3[2][2];
        for (int t = 0; t < 2; ++t) {
          int nt = wave * 2 + t;
          b3[t][0] = load_b_t(stage, HT, 0,  half, nt * 16 + l16);
          b3[t][1] = load_b_t(stage, HT, 32, half, nt * 16 + l16);
        }
#pragma unroll
        for (int dt = 0; dt < 4; ++dt) {
          v16bf a0 = load_a(hs, HT, dt * 16 + l16, 0, half);
          v16bf a1 = load_a(hs, HT, dt * 16 + l16, 32, half);
          for (int t = 0; t < 2; ++t) {
            zacc[dt][t] = wmma_bf16(a0, b3[t][0], zacc[dt][t]);
            zacc[dt][t] = wmma_bf16(a1, b3[t][1], zacc[dt][t]);
          }
        }
      }
    }  // chunk loop

    // ---- stage = z^T as bf16 [n][d] ----
    __syncthreads();
    for (int t = 0; t < 2; ++t) {
      int nt = wave * 2 + t;
#pragma unroll
      for (int dt = 0; dt < 4; ++dt)
#pragma unroll
        for (int i = 0; i < 8; ++i)
          stage[(nt * 16 + l16) * HD_ + dt * 16 + half * 8 + i] =
              (__bf16)zacc[dt][t][i];
    }
    __syncthreads();

    // ---- GEMM4: o = W2c(64x64) @ z + b2; weighted accumulate ----
    {
      v16bf zb[2][2];
      for (int t = 0; t < 2; ++t) {
        int nt = wave * 2 + t;
        zb[t][0] = load_b_t(stage, HD_, 0,  half, nt * 16 + l16);
        zb[t][1] = load_b_t(stage, HD_, 32, half, nt * 16 + l16);
      }
#pragma unroll
      for (int dt = 0; dt < 4; ++dt) {
        v16bf a0 = load_a(w2cs, HD_, dt * 16 + l16, 0, half);
        v16bf a1 = load_a(w2cs, HD_, dt * 16 + l16, 32, half);
        for (int t = 0; t < 2; ++t) {
          int nt = wave * 2 + t;
          v8f acc = v8zero();
          acc = wmma_bf16(a0, zb[t][0], acc);
          acc = wmma_bf16(a1, zb[t][1], acc);
#pragma unroll
          for (int i = 0; i < 8; ++i) {
            int m = dt * 16 + half * 8 + i, nn = nt * 16 + l16;
            outacc[dt][t][i] += wk * (acc[i] + b2g[(size_t)m * N_ + nn]);
          }
        }
      }
    }
    __syncthreads();  // protect shared buffers before next expert
  }  // k loop

  // ---- write output (B,H,64,256) f32 ----
  float* og = out + (size_t)bh * HD_ * N_;
  for (int t = 0; t < 2; ++t) {
    int nt = wave * 2 + t;
#pragma unroll
    for (int dt = 0; dt < 4; ++dt)
#pragma unroll
      for (int i = 0; i < 8; ++i) {
        int m = dt * 16 + half * 8 + i, nn = nt * 16 + l16;
        og[(size_t)m * N_ + nn] = outacc[dt][t][i];
      }
  }
}

extern "C" void kernel_launch(void* const* d_in, const int* in_sizes, int n_in,
                              void* d_out, int out_size, void* d_ws, size_t ws_size,
                              hipStream_t stream) {
  const float* x    = (const float*)d_in[0];
  const int*   eidx = (const int*)  d_in[1];
  const float* ewt  = (const float*)d_in[2];
  const float* w1t  = (const float*)d_in[3];
  const float* w1c  = (const float*)d_in[4];
  const float* b1   = (const float*)d_in[5];
  const float* w2t  = (const float*)d_in[6];
  const float* w2c  = (const float*)d_in[7];
  const float* b2   = (const float*)d_in[8];
  float* out = (float*)d_out;

  const size_t n_w1t = (size_t)E_ * H_ * HID_ * N_;   // 12.58M
  const size_t n_w2t = (size_t)E_ * H_ * N_ * HID_;   // 12.58M
  const size_t n_x   = (size_t)B_ * H_ * HD_ * N_;    //  6.29M
  const size_t need  = (n_w1t + n_w2t + n_x) * sizeof(__bf16);

  dim3 grid(B_ * H_);
  dim3 block(256);

  if (ws_size >= need) {
    __bf16* ws_w1t = (__bf16*)d_ws;
    __bf16* ws_w2t = ws_w1t + n_w1t;
    __bf16* ws_x   = ws_w2t + n_w2t;
    hipLaunchKernelGGL(cvt_bf16_kernel, dim3(2048), dim3(256), 0, stream,
                       w1t, ws_w1t, (int)(n_w1t / 4));
    hipLaunchKernelGGL(cvt_bf16_kernel, dim3(2048), dim3(256), 0, stream,
                       w2t, ws_w2t, (int)(n_w2t / 4));
    hipLaunchKernelGGL(cvt_bf16_kernel, dim3(1024), dim3(256), 0, stream,
                       x, ws_x, (int)(n_x / 4));
    hipLaunchKernelGGL((moe_mixer_kernel<true>), grid, block, 0, stream,
                       (const void*)ws_x, eidx, ewt, (const void*)ws_w1t, w1c,
                       b1, (const void*)ws_w2t, w2c, b2, out);
  } else {
    hipLaunchKernelGGL((moe_mixer_kernel<false>), grid, block, 0, stream,
                       (const void*)x, eidx, ewt, (const void*)w1t, w1c, b1,
                       (const void*)w2t, w2c, b2, out);
  }
}
